// ESIM_27582279975475
// MI455X (gfx1250) — matrix operand order repaired and checked by reading, not compiled, not run on hardware
//
#include <hip/hip_runtime.h>

// ---------------------------------------------------------------------------
// ESIM forward for MI455X (gfx1250, wave32, WMMA).
// - All GEMMs: v_wmma_f32_16x16x32_f16 with pre-packed f16 fragments
//   (A: [mt][kt][lane][16], B: [nt][kt][lane][16]) -> 32B b128 operand loads.
// - LSTM scans: one workgroup per run; h resident in LDS in A-fragment layout;
//   x fragments double-buffered into LDS with global_load_async_to_lds_b128
//   (ASYNCcnt / s_wait_asynccnt) so next-step input fetch overlaps the gate GEMM.
// ---------------------------------------------------------------------------

#define T1D 48
#define T2D 48
#define BZ  64
#define ED  300
#define HD  300
#define G4  1200            // 4*H
#define KT300 10            // ceil(300/32)
#define MT64  4             // 64/16
#define NT1200 75           // 1200/16
#define FRAG 512            // halves per fragment tile (32 lanes * 16)
#define XBLK (MT64 * KT300 * FRAG)   // 20480 halves = 40 KB per timestep

typedef __attribute__((ext_vector_type(16))) _Float16 v16h;
typedef __attribute__((ext_vector_type(8)))  float    v8f;

// A-matrix 16x32 f16 K offset (ISA 7.12.2): lanes 0-15 K={0..7,16..23}, lanes 16-31 +8
__device__ __forceinline__ int koffA(int j, int g) {
    return ((j < 8) ? j : (j + 8)) + 8 * g;
}

__device__ __forceinline__ v8f wmma_step(v16h a, v16h b, v8f c) {
    return __builtin_amdgcn_wmma_f32_16x16x32_f16(false, a, false, b, (short)0, c, false, false);
}

// Async-stage one 40KB x-fragment block into LDS (10 ops x 256 thr x 16B).
// VDST = LDS byte offset (low 32 bits of flat shared address), VADDR = global.
__device__ __forceinline__ void stage_async(const _Float16* __restrict__ gsrc,
                                            _Float16* lbuf, int tid) {
#pragma unroll 1
    for (int i = 0; i < 10; ++i) {
        size_t byteoff = ((size_t)i * 256 + tid) * 16;
        unsigned lds = (unsigned)(uintptr_t)lbuf + (unsigned)byteoff;
        unsigned long long ga = (unsigned long long)(uintptr_t)((const char*)gsrc + byteoff);
        asm volatile("global_load_async_to_lds_b128 %0, %1, off"
                     :: "v"(lds), "v"(ga) : "memory");
    }
}

__device__ __forceinline__ void wait_async_all() {
    asm volatile("s_wait_asynccnt 0x0" ::: "memory");
}

// ---------------------------------------------------------------------------
// Pack W [N,K] f32 row-major -> B fragments (W^T operand for X @ W^T).
// ---------------------------------------------------------------------------
__global__ void pack_wT_kernel(const float* __restrict__ W, _Float16* __restrict__ out,
                               int N, int K)
{
    const int KT = (K + 31) / 32, NT = (N + 15) / 16;
    int id = blockIdx.x * blockDim.x + threadIdx.x;
    if (id >= NT * KT * 32) return;
    const int lane = id & 31;
    const int kt   = (id >> 5) % KT;
    const int nt   = (id >> 5) / KT;
    const int n = nt * 16 + (lane & 15);
    const int g = lane >> 4;
    v16h v;
#pragma unroll
    for (int j = 0; j < 16; ++j) {
        int k = kt * 32 + koffA(j, g);          // B layout: same K order as A (koffA)
        v[j] = (_Float16)((n < N && k < K) ? W[(long)n * K + k] : 0.0f);
    }
    *(v16h*)(out + (size_t)id * 16) = v;
}

// ---------------------------------------------------------------------------
// Pack A-side matrices into A fragments.
// MODE 0: dense f32 [M,K].   MODE 1: embedding gather (ids flat [T*64]).
// MODE 2: fused ESIM features [ctx, att, ctx*att, ctx-att], K=2400.
// ---------------------------------------------------------------------------
template <int MODE>
__global__ void pack_a_kernel(const float* __restrict__ src, const int* __restrict__ ids,
                              const float* __restrict__ emb,
                              const float* __restrict__ ctx, const float* __restrict__ att,
                              _Float16* __restrict__ out, int M, int K, int Tdim)
{
    const int KT = (K + 31) / 32, MT = (M + 15) / 16;
    int id = blockIdx.x * blockDim.x + threadIdx.x;
    if (id >= MT * KT * 32) return;
    const int lane = id & 31;
    const int kt   = (id >> 5) % KT;
    const int mt   = (id >> 5) / KT;
    const int m = mt * 16 + (lane & 15);
    const int g = lane >> 4;
    v16h v;
#pragma unroll
    for (int j = 0; j < 16; ++j) {
        int k = kt * 32 + koffA(j, g);
        float f = 0.0f;
        if (m < M && k < K) {
            if (MODE == 0) {
                f = src[(long)m * K + k];
            } else if (MODE == 1) {
                f = emb[(long)ids[m] * K + k];
            } else {
                const int t = m >> 6, b = m & 63;
                const int d = k % 600, seg = k / 600;
                float cv = ctx[((long)t * 64 + b) * 600 + d];
                float av = att[((long)b * Tdim + t) * 600 + d];
                f = (seg == 0) ? cv : (seg == 1) ? av : (seg == 2) ? cv * av : cv - av;
            }
        }
        v[j] = (_Float16)f;
    }
    *(v16h*)(out + (size_t)id * 16) = v;
}

// ---------------------------------------------------------------------------
// LSTM scan. 256 thr = 8 wave32. Each wave owns gate column nt with 4
// accumulators (B fragment reused 4x). x double-buffered via async-to-LDS;
// h lives in LDS in fragment layout.  kt loops kept rolled (#pragma unroll 1)
// to avoid the register spills a full unroll causes.
// ---------------------------------------------------------------------------
__global__ void lstm_run_kernel(const _Float16* __restrict__ xpk,  // [T][4][10][512]
                                const _Float16* __restrict__ wip,  // [75][10][512]
                                const _Float16* __restrict__ whp,  // [75][10][512]
                                const float* __restrict__ bih, const float* __restrict__ bhh,
                                int T, int rev,
                                float* __restrict__ gates,         // [64*1200]
                                float* __restrict__ cbuf,          // [64*300]
                                float* __restrict__ hout)          // [64*300]
{
    __shared__ _Float16 hfr[XBLK];       // 40 KB: h in A-fragment layout
    __shared__ _Float16 xst[2][XBLK];    // 2 x 40 KB: async x staging
    const int tid = threadIdx.x, wave = tid >> 5, lane = tid & 31;

    for (int i = tid; i < XBLK; i += 256) hfr[i] = (_Float16)0.0f;
    for (int i = tid; i < BZ * HD; i += 256) cbuf[i] = 0.0f;
    __syncthreads();

    // prime buffer 0 with x(ts(0))
    stage_async(xpk + (size_t)(rev ? (T - 1) : 0) * XBLK, xst[0], tid);

    for (int t = 0; t < T; ++t) {
        wait_async_all();            // xst[t&1] landed (own wave's ops)
        __syncthreads();             // ... and everyone else's
        if (t + 1 < T) {             // overlap next-step fetch with this GEMM
            const int tsn = rev ? (T - 2 - t) : (t + 1);
            stage_async(xpk + (size_t)tsn * XBLK, xst[(t + 1) & 1], tid);
        }
        const _Float16* xt = xst[t & 1];

        for (int nt = wave; nt < NT1200; nt += 8) {
            v8f acc0 = {0,0,0,0,0,0,0,0}, acc1 = {0,0,0,0,0,0,0,0};
            v8f acc2 = {0,0,0,0,0,0,0,0}, acc3 = {0,0,0,0,0,0,0,0};
            // x_t @ Wih^T
#pragma unroll 1
            for (int kt = 0; kt < KT300; ++kt) {
                v16h b = *(const v16h*)(wip + ((size_t)(nt * KT300 + kt) * 32 + lane) * 16);
                int ktn = (kt < KT300 - 1) ? kt + 1 : kt;
                __builtin_prefetch(wip + ((size_t)(nt * KT300 + ktn) * 32 + lane) * 16, 0, 3);
                v16h a0 = *(const v16h*)(xt + ((0 * KT300 + kt) * 32 + lane) * 16);
                v16h a1 = *(const v16h*)(xt + ((1 * KT300 + kt) * 32 + lane) * 16);
                v16h a2 = *(const v16h*)(xt + ((2 * KT300 + kt) * 32 + lane) * 16);
                v16h a3 = *(const v16h*)(xt + ((3 * KT300 + kt) * 32 + lane) * 16);
                acc0 = wmma_step(a0, b, acc0);
                acc1 = wmma_step(a1, b, acc1);
                acc2 = wmma_step(a2, b, acc2);
                acc3 = wmma_step(a3, b, acc3);
            }
            // h @ Whh^T
#pragma unroll 1
            for (int kt = 0; kt < KT300; ++kt) {
                v16h b = *(const v16h*)(whp + ((size_t)(nt * KT300 + kt) * 32 + lane) * 16);
                v16h a0 = *(const v16h*)(hfr + ((0 * KT300 + kt) * 32 + lane) * 16);
                v16h a1 = *(const v16h*)(hfr + ((1 * KT300 + kt) * 32 + lane) * 16);
                v16h a2 = *(const v16h*)(hfr + ((2 * KT300 + kt) * 32 + lane) * 16);
                v16h a3 = *(const v16h*)(hfr + ((3 * KT300 + kt) * 32 + lane) * 16);
                acc0 = wmma_step(a0, b, acc0);
                acc1 = wmma_step(a1, b, acc1);
                acc2 = wmma_step(a2, b, acc2);
                acc3 = wmma_step(a3, b, acc3);
            }
            const int n = nt * 16 + (lane & 15);
            const int g = lane >> 4;
#pragma unroll
            for (int r = 0; r < 8; ++r) gates[(long)(g * 8 + r)      * G4 + n] = acc0[r];
#pragma unroll
            for (int r = 0; r < 8; ++r) gates[(long)(16 + g * 8 + r) * G4 + n] = acc1[r];
#pragma unroll
            for (int r = 0; r < 8; ++r) gates[(long)(32 + g * 8 + r) * G4 + n] = acc2[r];
#pragma unroll
            for (int r = 0; r < 8; ++r) gates[(long)(48 + g * 8 + r) * G4 + n] = acc3[r];
        }
        __threadfence();
        __syncthreads();

        // ---- cell update (gate order i,f,g,o); h written back in frag layout
        for (int e = tid; e < BZ * HD; e += 256) {
            const int m = e / HD, kk = e % HD;
            const long rb = (long)m * G4;
            float gi = gates[rb + kk]        + bih[kk]        + bhh[kk];
            float gf = gates[rb + kk + HD]   + bih[kk + HD]   + bhh[kk + HD];
            float gg = gates[rb + kk + 2*HD] + bih[kk + 2*HD] + bhh[kk + 2*HD];
            float go = gates[rb + kk + 3*HD] + bih[kk + 3*HD] + bhh[kk + 3*HD];
            float si = 1.0f / (1.0f + __expf(-gi));
            float sf = 1.0f / (1.0f + __expf(-gf));
            float so = 1.0f / (1.0f + __expf(-go));
            float c  = sf * cbuf[e] + si * tanhf(gg);
            cbuf[e]  = c;
            float h  = so * tanhf(c);
            const int mt = m >> 4, kt = kk >> 5, ko = kk & 31;
            int g2, jj;
            if (ko < 8)       { g2 = 0; jj = ko; }
            else if (ko < 16) { g2 = 1; jj = ko - 8; }
            else if (ko < 24) { g2 = 0; jj = ko - 8; }
            else              { g2 = 1; jj = ko - 16; }
            hfr[((mt * KT300 + kt) * 32 + (g2 * 16 + (m & 15))) * 16 + jj] = (_Float16)h;
            if (t == T - 1) hout[e] = h;
        }
        __syncthreads();
    }
}

// ---------------------------------------------------------------------------
// Packed-operand GEMM: C[M,N] = act(A @ W^T + bias). One wave per tile.
// ---------------------------------------------------------------------------
__global__ void gemm_packed_kernel(const _Float16* __restrict__ Apk,
                                   const _Float16* __restrict__ Wp,
                                   const float* __restrict__ bias,
                                   float* __restrict__ C,
                                   int M, int N, int KT, int act)
{
    const int NT = (N + 15) / 16, MT = (M + 15) / 16;
    const int ntiles = MT * NT;
    const int lane   = threadIdx.x & 31;
    const int wglob  = (blockIdx.x * blockDim.x + threadIdx.x) >> 5;
    const int nwaves = (gridDim.x * blockDim.x) >> 5;

    for (int tile = wglob; tile < ntiles; tile += nwaves) {
        const int mt = tile / NT, nt = tile % NT;
        v8f acc = {0,0,0,0,0,0,0,0};
#pragma unroll 1
        for (int kt = 0; kt < KT; ++kt) {
            v16h a = *(const v16h*)(Apk + ((size_t)(mt * KT + kt) * 32 + lane) * 16);
            v16h b = *(const v16h*)(Wp  + ((size_t)(nt * KT + kt) * 32 + lane) * 16);
            acc = wmma_step(a, b, acc);
        }
        const int n = nt * 16 + (lane & 15);
        const int g = lane >> 4;
        if (n < N) {
            const float bv = bias ? bias[n] : 0.0f;
#pragma unroll
            for (int r = 0; r < 8; ++r) {
                int mm = mt * 16 + g * 8 + r;
                if (mm < M) {
                    float v = acc[r] + bv;
                    if (act == 1)      v = fmaxf(v, 0.0f);
                    else if (act == 2) v = tanhf(v);
                    C[(long)mm * N + n] = v;
                }
            }
        }
    }
}

// ---- encoder context: ctx[t,b,:] = concat(hf[b], hr[63-b]) * mask[t,b]
__global__ void ctx_kernel(const float* __restrict__ hf, const float* __restrict__ hr,
                           const float* __restrict__ mask, float* __restrict__ ctx, int T)
{
    int i = blockIdx.x * blockDim.x + threadIdx.x;
    if (i >= T * BZ * 600) return;
    int d = i % 600, b = (i / 600) % BZ, t = i / (600 * BZ);
    float v = (d < 300) ? hf[b * 300 + d] : hr[(63 - b) * 300 + (d - 300)];
    ctx[i] = v * mask[t * BZ + b];
}

// ---- scores[b,i,j] = max_d c1[b,i,d]*c2[b,j,d]  (max-product: VALU only)
__global__ void scores_kernel(const float* __restrict__ ctx1,
                              const float* __restrict__ ctx2,
                              float* __restrict__ scores)
{
    int id = blockIdx.x * blockDim.x + threadIdx.x;
    if (id >= BZ * T1D * T2D) return;
    int j = id % T2D, i = (id / T2D) % T1D, b = id / (T2D * T1D);
    const float* p1 = ctx1 + ((long)i * BZ + b) * 600;
    const float* p2 = ctx2 + ((long)j * BZ + b) * 600;
    float mx = -3.402823e38f;
    for (int d = 0; d < 600; ++d) mx = fmaxf(mx, p1[d] * p2[d]);
    scores[((long)b * T1D + i) * T2D + j] = mx;
}

__global__ void softmax_j_kernel(const float* __restrict__ scores,
                                 const float* __restrict__ mask2,
                                 float* __restrict__ alpha)
{
    int id = blockIdx.x * blockDim.x + threadIdx.x;
    if (id >= BZ * T1D) return;
    int i = id % T1D, b = id / T1D;
    const float* s = scores + ((long)b * T1D + i) * T2D;
    float mx = -3.402823e38f;
    for (int j = 0; j < T2D; ++j) mx = fmaxf(mx, s[j]);
    float e[T2D], sum = 0.0f;
    for (int j = 0; j < T2D; ++j) { float v = __expf(s[j] - mx) * mask2[j * BZ + b]; e[j] = v; sum += v; }
    for (int j = 0; j < T2D; ++j) alpha[((long)b * T1D + i) * T2D + j] = e[j] / sum;
}

// ---- a_tilde[b,i,d] = max_j alpha[b,i,j]*c2[b,j,d]  (faithful quirk: MAX pool)
__global__ void atilde_kernel(const float* __restrict__ alpha,
                              const float* __restrict__ ctx2,
                              float* __restrict__ at)
{
    int id = blockIdx.x * blockDim.x + threadIdx.x;
    if (id >= BZ * T1D * 600) return;
    int d = id % 600, i = (id / 600) % T1D, b = id / (600 * T1D);
    float mx = -3.402823e38f;
    for (int j = 0; j < T2D; ++j) {
        float v = alpha[((long)b * T1D + i) * T2D + j] * ctx2[((long)j * BZ + b) * 600 + d];
        mx = fmaxf(mx, v);
    }
    at[((long)b * T1D + i) * 600 + d] = mx;   // [B,T1,600]
}

__global__ void softmax_i_kernel(const float* __restrict__ scores,
                                 const float* __restrict__ mask1,
                                 float* __restrict__ beta)
{
    int id = blockIdx.x * blockDim.x + threadIdx.x;
    if (id >= BZ * T2D) return;
    int j = id % T2D, b = id / T2D;
    float mx = -3.402823e38f;
    for (int i = 0; i < T1D; ++i) mx = fmaxf(mx, scores[((long)b * T1D + i) * T2D + j]);
    float e[T1D], sum = 0.0f;
    for (int i = 0; i < T1D; ++i) {
        float v = __expf(scores[((long)b * T1D + i) * T2D + j] - mx) * mask1[i * BZ + b];
        e[i] = v; sum += v;
    }
    for (int i = 0; i < T1D; ++i) beta[((long)b * T1D + i) * T2D + j] = e[i] / sum;
}

// ---- b_tilde[b,j,d] = sum_i beta[b,i,j]*c1[b,i,d]
__global__ void btilde_kernel(const float* __restrict__ beta,
                              const float* __restrict__ ctx1,
                              float* __restrict__ bt)
{
    int id = blockIdx.x * blockDim.x + threadIdx.x;
    if (id >= BZ * T2D * 600) return;
    int d = id % 600, j = (id / 600) % T2D, b = id / (600 * T2D);
    float s = 0.0f;
    for (int i = 0; i < T1D; ++i)
        s += beta[((long)b * T1D + i) * T2D + j] * ctx1[((long)i * BZ + b) * 600 + d];
    bt[((long)b * T2D + j) * 600 + d] = s;    // [B,T2,600]
}

// ---- pooled features: [mean(w3), max(w3), mean(w4), max(w4)] -> [B,2400]
__global__ void feat_kernel(const float* __restrict__ h3f, const float* __restrict__ h3r,
                            const float* __restrict__ h4f, const float* __restrict__ h4r,
                            const float* __restrict__ mask1, const float* __restrict__ mask2,
                            float* __restrict__ feat)
{
    int id = blockIdx.x * blockDim.x + threadIdx.x;
    if (id >= BZ * 2400) return;
    int col = id % 2400, b = id / 2400;
    int d = col % 600, seg = col / 600;
    float c3 = (d < 300) ? h3f[b * 300 + d] : h3r[(63 - b) * 300 + (d - 300)];
    float c4 = (d < 300) ? h4f[b * 300 + d] : h4r[(63 - b) * 300 + (d - 300)];
    float v;
    if (seg == 0) {
        float s = 0.f, ms = 0.f;
        for (int t = 0; t < T1D; ++t) { float m = mask1[t * BZ + b]; s += c3 * m; ms += m; }
        v = s / ms;
    } else if (seg == 1) {
        float mx = -3.402823e38f;
        for (int t = 0; t < T1D; ++t) mx = fmaxf(mx, c3 * mask1[t * BZ + b]);
        v = mx;
    } else if (seg == 2) {
        float s = 0.f, ms = 0.f;
        for (int t = 0; t < T2D; ++t) { float m = mask2[t * BZ + b]; s += c4 * m; ms += m; }
        v = s / ms;
    } else {
        float mx = -3.402823e38f;
        for (int t = 0; t < T2D; ++t) mx = fmaxf(mx, c4 * mask2[t * BZ + b]);
        v = mx;
    }
    feat[(long)b * 2400 + col] = v;
}

__global__ void final_kernel(const float* __restrict__ logit2,
                             const float* __restrict__ fow,
                             const float* __restrict__ fob,
                             float* __restrict__ out)
{
    int id = blockIdx.x * blockDim.x + threadIdx.x;
    if (id >= BZ * 3) return;
    int c = id % 3, b = id / 3;
    float s = fob[c];
    for (int d = 0; d < 300; ++d) s += logit2[b * 300 + d] * fow[c * 300 + d];
    out[b * 3 + c] = s;
}

// ---------------------------------------------------------------------------
extern "C" void kernel_launch(void* const* d_in, const int* in_sizes, int n_in,
                              void* d_out, int out_size, void* d_ws, size_t ws_size,
                              hipStream_t stream) {
    (void)in_sizes; (void)n_in; (void)out_size; (void)ws_size;
    const int*   x1      = (const int*)  d_in[0];
    const float* x1_mask = (const float*)d_in[1];
    const int*   x2      = (const int*)  d_in[2];
    const float* x2_mask = (const float*)d_in[3];
    const float* emb     = (const float*)d_in[5];
    const float* eW  = (const float*)d_in[6],  *eU  = (const float*)d_in[7];
    const float* ebi = (const float*)d_in[8],  *ebh = (const float*)d_in[9];
    const float* rW  = (const float*)d_in[10], *rU  = (const float*)d_in[11];
    const float* rbi = (const float*)d_in[12], *rbh = (const float*)d_in[13];
    const float* dW  = (const float*)d_in[14], *dU  = (const float*)d_in[15];
    const float* dbi = (const float*)d_in[16], *dbh = (const float*)d_in[17];
    const float* sW  = (const float*)d_in[18], *sU  = (const float*)d_in[19];
    const float* sbi = (const float*)d_in[20], *sbh = (const float*)d_in[21];
    const float* f1w = (const float*)d_in[22], *f1b = (const float*)d_in[23];
    const float* f2w = (const float*)d_in[24], *f2b = (const float*)d_in[25];
    const float* fow = (const float*)d_in[26], *fob = (const float*)d_in[27];
    float* out = (float*)d_out;

    // --- workspace arena (~72 MB) ---
    char* base = (char*)d_ws;
    size_t off = 0;
    auto alloc_f = [&](size_t n) { float* p = (float*)(base + off);
                                   off = (off + n * 4 + 255) & ~(size_t)255; return p; };
    auto alloc_h = [&](size_t n) { _Float16* p = (_Float16*)(base + off);
                                   off = (off + n * 2 + 255) & ~(size_t)255; return p; };

    float* gates = alloc_f(BZ * G4);
    float* cbuf  = alloc_f(BZ * HD);
    float* h1f = alloc_f(BZ * HD), *h1r = alloc_f(BZ * HD);
    float* h2f = alloc_f(BZ * HD), *h2r = alloc_f(BZ * HD);
    float* ctx1 = alloc_f((size_t)T1D * BZ * 600);
    float* ctx2 = alloc_f((size_t)T2D * BZ * 600);
    float* scores = alloc_f((size_t)BZ * T1D * T2D);
    float* alpha  = alloc_f((size_t)BZ * T1D * T2D);
    float* beta   = alloc_f((size_t)BZ * T1D * T2D);
    float* at = alloc_f((size_t)BZ * T1D * 600);
    float* bt = alloc_f((size_t)BZ * T2D * 600);
    float* h1seq = alloc_f((size_t)T1D * BZ * HD);
    float* h2seq = alloc_f((size_t)T2D * BZ * HD);
    float* hd1f = alloc_f(BZ * HD), *hd1r = alloc_f(BZ * HD);
    float* hd2f = alloc_f(BZ * HD), *hd2r = alloc_f(BZ * HD);
    float* feat   = alloc_f((size_t)BZ * 2400);
    float* logit2 = alloc_f((size_t)BZ * HD);

    const size_t WLSTM = (size_t)NT1200 * KT300 * FRAG;     // 384000 halves
    _Float16* eWp = alloc_h(WLSTM), *eUp = alloc_h(WLSTM);
    _Float16* rWp = alloc_h(WLSTM), *rUp = alloc_h(WLSTM);
    _Float16* dWp = alloc_h(WLSTM), *dUp = alloc_h(WLSTM);
    _Float16* sWp = alloc_h(WLSTM), *sUp = alloc_h(WLSTM);
    const size_t WFC = (size_t)19 * 75 * FRAG;              // 729600 halves
    _Float16* f1wp = alloc_h(WFC), *f2wp = alloc_h(WFC);
    const size_t SEQPK = (size_t)T1D * XBLK;                // 983040 halves
    _Float16* x1pk = alloc_h(SEQPK), *x2pk = alloc_h(SEQPK);
    _Float16* h1pk = alloc_h(SEQPK), *h2pk = alloc_h(SEQPK);
    _Float16* inppk  = alloc_h((size_t)192 * 75 * FRAG);    // reused for seq1/seq2
    _Float16* featpk = alloc_h((size_t)MT64 * 75 * FRAG);

    const int TPB = 256;
    auto blk = [](long n) { return (int)((n + 255) / 256); };

    // --- pack weights into B-fragment layout (once per launch, L2-resident) ---
    const long wl = (long)NT1200 * KT300 * 32;
    pack_wT_kernel<<<blk(wl), TPB, 0, stream>>>(eW, eWp, G4, ED);
    pack_wT_kernel<<<blk(wl), TPB, 0, stream>>>(eU, eUp, G4, HD);
    pack_wT_kernel<<<blk(wl), TPB, 0, stream>>>(rW, rWp, G4, ED);
    pack_wT_kernel<<<blk(wl), TPB, 0, stream>>>(rU, rUp, G4, HD);
    pack_wT_kernel<<<blk(wl), TPB, 0, stream>>>(dW, dWp, G4, HD);
    pack_wT_kernel<<<blk(wl), TPB, 0, stream>>>(dU, dUp, G4, HD);
    pack_wT_kernel<<<blk(wl), TPB, 0, stream>>>(sW, sWp, G4, HD);
    pack_wT_kernel<<<blk(wl), TPB, 0, stream>>>(sU, sUp, G4, HD);
    const long wf = (long)19 * 75 * 32;
    pack_wT_kernel<<<blk(wf), TPB, 0, stream>>>(f1w, f1wp, HD, 2400);
    pack_wT_kernel<<<blk(wf), TPB, 0, stream>>>(f2w, f2wp, HD, 2400);

    // --- pack embedding sequences into A-fragment layout ---
    const long sl = (long)192 * KT300 * 32;                 // M=3072, K=300
    pack_a_kernel<1><<<blk(sl), TPB, 0, stream>>>(nullptr, x1, emb, nullptr, nullptr,
                                                  x1pk, T1D * BZ, ED, 0);
    pack_a_kernel<1><<<blk(sl), TPB, 0, stream>>>(nullptr, x2, emb, nullptr, nullptr,
                                                  x2pk, T2D * BZ, ED, 0);

    // --- encoder BiLSTMs ---
    lstm_run_kernel<<<1, TPB, 0, stream>>>(x1pk, eWp, eUp, ebi, ebh, T1D, 0, gates, cbuf, h1f);
    lstm_run_kernel<<<1, TPB, 0, stream>>>(x1pk, rWp, rUp, rbi, rbh, T1D, 1, gates, cbuf, h1r);
    lstm_run_kernel<<<1, TPB, 0, stream>>>(x2pk, eWp, eUp, ebi, ebh, T2D, 0, gates, cbuf, h2f);
    lstm_run_kernel<<<1, TPB, 0, stream>>>(x2pk, rWp, rUp, rbi, rbh, T2D, 1, gates, cbuf, h2r);

    ctx_kernel<<<blk((long)T1D * BZ * 600), TPB, 0, stream>>>(h1f, h1r, x1_mask, ctx1, T1D);
    ctx_kernel<<<blk((long)T2D * BZ * 600), TPB, 0, stream>>>(h2f, h2r, x2_mask, ctx2, T2D);

    // --- attention ---
    scores_kernel<<<blk((long)BZ * T1D * T2D), TPB, 0, stream>>>(ctx1, ctx2, scores);
    softmax_j_kernel<<<blk(BZ * T1D), TPB, 0, stream>>>(scores, x2_mask, alpha);
    atilde_kernel<<<blk((long)BZ * T1D * 600), TPB, 0, stream>>>(alpha, ctx2, at);
    softmax_i_kernel<<<blk(BZ * T2D), TPB, 0, stream>>>(scores, x1_mask, beta);
    btilde_kernel<<<blk((long)BZ * T2D * 600), TPB, 0, stream>>>(beta, ctx1, bt);

    // --- fc1: h = relu([c,a,c*a,c-a] @ f1w^T + f1b); pack buffer reused ---
    {
        const long pl = (long)192 * 75 * 32;                // M=3072, K=2400
        int ntiles = 192 * 19, nb = (ntiles + 7) / 8;
        pack_a_kernel<2><<<blk(pl), TPB, 0, stream>>>(nullptr, nullptr, nullptr, ctx1, at,
                                                      inppk, T1D * BZ, 2400, T1D);
        gemm_packed_kernel<<<nb, TPB, 0, stream>>>(inppk, f1wp, f1b, h1seq, T1D * BZ, HD, 75, 1);
        pack_a_kernel<2><<<blk(pl), TPB, 0, stream>>>(nullptr, nullptr, nullptr, ctx2, bt,
                                                      inppk, T2D * BZ, 2400, T2D);
        gemm_packed_kernel<<<nb, TPB, 0, stream>>>(inppk, f1wp, f1b, h2seq, T2D * BZ, HD, 75, 1);
    }

    // --- decoder BiLSTMs ---
    pack_a_kernel<0><<<blk(sl), TPB, 0, stream>>>(h1seq, nullptr, nullptr, nullptr, nullptr,
                                                  h1pk, T1D * BZ, HD, 0);
    pack_a_kernel<0><<<blk(sl), TPB, 0, stream>>>(h2seq, nullptr, nullptr, nullptr, nullptr,
                                                  h2pk, T2D * BZ, HD, 0);
    lstm_run_kernel<<<1, TPB, 0, stream>>>(h1pk, dWp, dUp, dbi, dbh, T1D, 0, gates, cbuf, hd1f);
    lstm_run_kernel<<<1, TPB, 0, stream>>>(h1pk, sWp, sUp, sbi, sbh, T1D, 1, gates, cbuf, hd1r);
    lstm_run_kernel<<<1, TPB, 0, stream>>>(h2pk, dWp, dUp, dbi, dbh, T2D, 0, gates, cbuf, hd2f);
    lstm_run_kernel<<<1, TPB, 0, stream>>>(h2pk, sWp, sUp, sbi, sbh, T2D, 1, gates, cbuf, hd2r);

    // --- pooling + fc2 + output ---
    feat_kernel<<<blk((long)BZ * 2400), TPB, 0, stream>>>(hd1f, hd1r, hd2f, hd2r,
                                                          x1_mask, x2_mask, feat);
    {
        const long pl = (long)MT64 * 75 * 32;               // M=64, K=2400
        pack_a_kernel<0><<<blk(pl), TPB, 0, stream>>>(feat, nullptr, nullptr, nullptr, nullptr,
                                                      featpk, BZ, 2400, 0);
        int ntiles = MT64 * 19, nb = (ntiles + 7) / 8;
        gemm_packed_kernel<<<nb, TPB, 0, stream>>>(featpk, f2wp, f2b, logit2, BZ, HD, 75, 2);
    }
    final_kernel<<<1, TPB, 0, stream>>>(logit2, fow, fob, out);
}
